// SwinTransformer_9259949490813
// MI455X (gfx1250) — matrix operand order repaired and checked by reading, not compiled, not run on hardware
//
#include <hip/hip_runtime.h>
#include <hip/hip_bf16.h>
#include <cstdint>
#include <cstddef>

// ---------------------------------------------------------------------------
// Swin block: LN1 -> shifted-window MHSA (+rel-pos bias, +shift mask) -> proj
//             -> residual -> LN2 -> MLP(GELU) -> residual
// fp32 in/out, f16 WMMA math with f32 accumulation (v_wmma_f32_16x16x32_f16).
// GEMM tiles are fed by GLOBAL_LOAD_ASYNC_TO_LDS_B128 with double buffering
// (ASYNCcnt / s_wait_asynccnt), weights pre-transposed so no LDS transpose.
// ---------------------------------------------------------------------------

#define DIMC   384
#define NHEAD  12
#define HD     32
#define WSZ    7
#define NTOK   49          // tokens per window
#define IMGH   56
#define NWIN   64          // windows per image (8x8)
#define TOK    100352      // 32*3136 == 2048*49 windowed rows
#define NMLP   1536

typedef __attribute__((ext_vector_type(16))) _Float16 v16h;
typedef __attribute__((ext_vector_type(8)))  _Float16 v8h;
typedef __attribute__((ext_vector_type(8)))  float    v8f;

// -------------------------------- utilities --------------------------------

// weight convert + transpose: out[N][K] (f16) = in[K][N] (f32)^T
__global__ void k_cvt_t_f16(const float* __restrict__ in, _Float16* __restrict__ out,
                            int K, int N) {
  int i = blockIdx.x * 256 + threadIdx.x;
  if (i < K * N) {
    const int n = i / K, k = i - n * K;
    out[i] = (_Float16)in[(size_t)k * N + n];
  }
}

__device__ __forceinline__ int swin_region(int x) {
  // slices on rolled image: [0,49) [49,53) [53,56)
  return x < 49 ? 0 : (x < 53 ? 1 : 2);
}

// async copy of a 128x32-half tile (row stride ldk halfs) into LDS.
// 512 x 16B chunks over 256 threads -> 2 async instructions per wave / tile.
__device__ __forceinline__ void async_tile_128x32(const _Float16* __restrict__ gsrc,
                                                  _Float16* ldst, int ldk, int tid) {
  #pragma unroll
  for (int i = tid; i < 512; i += 256) {
    const int row = i >> 2, seg = (i & 3) * 8;
    const uint32_t lds = (uint32_t)(uintptr_t)(ldst + row * 32 + seg);
    const uint64_t g   = (uint64_t)(uintptr_t)(gsrc + (size_t)row * ldk + seg);
    asm volatile("global_load_async_to_lds_b128 %0, %1, off"
                 :: "v"(lds), "v"(g) : "memory");
  }
}

// ---------------- LN1 + roll(-3,-3) + window partition (gather) -------------
// one wave per destination windowed row; each lane owns 12 channels
__global__ __launch_bounds__(256) void k_ln1_window(
    const float* __restrict__ x, const float* __restrict__ g,
    const float* __restrict__ be, _Float16* __restrict__ xw) {
  const int lane = threadIdx.x & 31, wave = threadIdx.x >> 5;
  const int r = blockIdx.x * 8 + wave;          // windowed row
  if (r >= TOK) return;
  const int widx = r / NTOK, n = r - widx * NTOK;
  const int b = widx >> 6, wi = widx & 63;
  const int wr = wi >> 3, wc = wi & 7;
  const int hp = wr * WSZ + n / WSZ;            // rolled coords
  const int wp = wc * WSZ + n % WSZ;
  int hh = hp + 3; if (hh >= IMGH) hh -= IMGH;  // rolled[hp] = x[(hp+3)%56]
  int ww = wp + 3; if (ww >= IMGH) ww -= IMGH;
  const float* row = x + ((size_t)b * 3136 + hh * IMGH + ww) * DIMC;
  float v[12], s = 0.f, s2 = 0.f;
  #pragma unroll
  for (int i = 0; i < 12; ++i) {
    float t = row[lane + i * 32]; v[i] = t; s += t; s2 += t * t;
  }
  #pragma unroll
  for (int m = 16; m >= 1; m >>= 1) { s += __shfl_xor(s, m, 32); s2 += __shfl_xor(s2, m, 32); }
  const float mu = s * (1.f / DIMC);
  const float rstd = rsqrtf(s2 * (1.f / DIMC) - mu * mu + 1e-5f);
  _Float16* o = xw + (size_t)r * DIMC;
  #pragma unroll
  for (int i = 0; i < 12; ++i) {
    const int c = lane + i * 32;
    o[c] = (_Float16)((v[i] - mu) * rstd * g[c] + be[c]);
  }
}

// ------------- window reverse + roll(+3,+3) + residual + LN2 ----------------
__global__ __launch_bounds__(256) void k_unwin_ln2(
    const float* __restrict__ ywin, const float* __restrict__ x,
    const float* __restrict__ g2, const float* __restrict__ be2,
    float* __restrict__ x1, _Float16* __restrict__ ln2) {
  const int lane = threadIdx.x & 31, wave = threadIdx.x >> 5;
  const int r = blockIdx.x * 8 + wave;          // (b, l) token
  if (r >= TOK) return;
  const int b = r / 3136, l = r - b * 3136;
  const int hh = l / IMGH, ww = l - hh * IMGH;
  int hp = hh - 3; if (hp < 0) hp += IMGH;      // final[hh] = out_rolled[(hh-3)%56]
  int wp = ww - 3; if (wp < 0) wp += IMGH;
  const int wr = hp / WSZ, nh = hp - wr * WSZ;
  const int wc = wp / WSZ, nw = wp - wc * WSZ;
  const size_t src = (((size_t)b * NWIN + wr * 8 + wc) * NTOK) + nh * WSZ + nw;
  const float* yr  = ywin + src * DIMC;
  const float* xr  = x + (size_t)r * DIMC;
  float* x1r = x1 + (size_t)r * DIMC;
  float v[12], s = 0.f, s2 = 0.f;
  #pragma unroll
  for (int i = 0; i < 12; ++i) {
    const int c = lane + i * 32;
    float t = yr[c] + xr[c];
    v[i] = t; s += t; s2 += t * t;
    x1r[c] = t;
  }
  #pragma unroll
  for (int m = 16; m >= 1; m >>= 1) { s += __shfl_xor(s, m, 32); s2 += __shfl_xor(s2, m, 32); }
  const float mu = s * (1.f / DIMC);
  const float rstd = rsqrtf(s2 * (1.f / DIMC) - mu * mu + 1e-5f);
  _Float16* lr = ln2 + (size_t)r * DIMC;
  #pragma unroll
  for (int i = 0; i < 12; ++i) {
    const int c = lane + i * 32;
    lr[c] = (_Float16)((v[i] - mu) * rstd * g2[c] + be2[c]);
  }
}

// ---------------------- generic 128x128 WMMA GEMM ---------------------------
// C[M,N] = A[M,K](f16) * Bt[N,K](f16)^T + bias ; optional GELU/residual/f32 out.
// 256 threads = 8 waves in 4x2 grid; wave tile 32x64 = 2x4 accumulators.
// Double-buffered LDS, tiles DMA'd with global_load_async_to_lds_b128.
template<bool OUTF32, bool DOGELU, bool DORESID>
__global__ __launch_bounds__(256) void k_gemm(
    const _Float16* __restrict__ A, const _Float16* __restrict__ Bt,
    const float* __restrict__ bias, const float* __restrict__ resid,
    void* __restrict__ outp, int M, int Ndim, int K) {
  const int t = threadIdx.x;
  const int lane = t & 31, wv = t >> 5;
  const int wm = wv >> 1, wn = wv & 1;
  const int mbase = blockIdx.y * 128;
  const int nbase = blockIdx.x * 128;

  __shared__ _Float16 sA[2][128 * 32];   // [m][k]
  __shared__ _Float16 sB[2][128 * 32];   // [n][k]

  v8f acc[2][4];
  #pragma unroll
  for (int mi = 0; mi < 2; ++mi)
    #pragma unroll
    for (int ni = 0; ni < 4; ++ni) { v8f z = {}; acc[mi][ni] = z; }

  const _Float16* Abase = A  + (size_t)mbase * K;
  const _Float16* Bbase = Bt + (size_t)nbase * K;
  const int kTiles = K >> 5;

  // prologue: tile 0 into buffer 0 (4 async instructions / wave)
  async_tile_128x32(Abase, &sA[0][0], K, t);
  async_tile_128x32(Bbase, &sB[0][0], K, t);

  for (int kt = 0; kt < kTiles; ++kt) {
    const int cur = kt & 1;
    if (kt + 1 < kTiles) {
      const int kk = (kt + 1) << 5;
      async_tile_128x32(Abase + kk, &sA[cur ^ 1][0], K, t);
      async_tile_128x32(Bbase + kk, &sB[cur ^ 1][0], K, t);
      asm volatile("s_wait_asynccnt 4" ::: "memory");  // tile kt landed
    } else {
      asm volatile("s_wait_asynccnt 0" ::: "memory");
    }
    __syncthreads();  // all waves' DMA for tile kt visible

    v16h af[2];
    #pragma unroll
    for (int mi = 0; mi < 2; ++mi) {
      const int mrow = wm * 32 + mi * 16 + (lane & 15);
      const int kh = (lane >> 4) * 16;
      v8h lo = *(const v8h*)&sA[cur][mrow * 32 + kh];
      v8h hi = *(const v8h*)&sA[cur][mrow * 32 + 8 + kh];
      af[mi] = __builtin_shufflevector(lo, hi, 0,1,2,3,4,5,6,7,8,9,10,11,12,13,14,15);
    }
    #pragma unroll
    for (int ni = 0; ni < 4; ++ni) {
      const int ncol = wn * 64 + ni * 16 + (lane & 15);
      const int kb = (lane >> 4) * 16;
      v16h bf = *(const v16h*)&sB[cur][ncol * 32 + kb];
      #pragma unroll
      for (int mi = 0; mi < 2; ++mi)
        acc[mi][ni] = __builtin_amdgcn_wmma_f32_16x16x32_f16(
            false, af[mi], false, bf, (short)0, acc[mi][ni], false, false);
    }
    __syncthreads();  // done reading buffer `cur` before it is re-filled
  }

  // epilogue
  #pragma unroll
  for (int mi = 0; mi < 2; ++mi)
    #pragma unroll
    for (int ni = 0; ni < 4; ++ni) {
      const int col = nbase + wn * 64 + ni * 16 + (lane & 15);
      const float bb = bias[col];
      #pragma unroll
      for (int v = 0; v < 8; ++v) {
        const int row = mbase + wm * 32 + mi * 16 + (lane >> 4) * 8 + v;
        float val = acc[mi][ni][v] + bb;
        if (DOGELU) val = 0.5f * val * (1.0f + erff(val * 0.70710678f));
        if (DORESID) val += resid[(size_t)row * Ndim + col];
        if (OUTF32) ((float*)outp)[(size_t)row * Ndim + col] = val;
        else        ((_Float16*)outp)[(size_t)row * Ndim + col] = (_Float16)val;
      }
    }
}

// --------------------------- windowed attention -----------------------------
// one 64-thread block (2 waves) per (window, head). 49 tokens padded to 64.
__global__ __launch_bounds__(64) void k_attn(
    const _Float16* __restrict__ qkv, const float* __restrict__ rel_table,
    _Float16* __restrict__ attnout) {
  const int bh = blockIdx.x;
  const int w  = bh / NHEAD;
  const int h  = bh - w * NHEAD;
  const int wi = w & 63;                  // window index within image (mask)
  const size_t wbase = (size_t)w * NTOK;
  const int t = threadIdx.x, lane = t & 31, wave = t >> 5;

  __shared__ _Float16 sQ[64 * 32];        // [tok][hd]
  __shared__ _Float16 sK[64 * 32];        // [tok][hd]
  __shared__ _Float16 sVt[32 * 64];       // [hd][tok]  (V transposed)
  __shared__ float    sS[64 * 64];        // scores / logits / exp
  __shared__ _Float16 sP[64 * 64];        // softmax probabilities (f16)

  // ---- load q/k/v rows (row = thread id; pad rows/cols with zero) ----
  {
    const _Float16* qb = qkv + wbase * (3 * DIMC) + h * HD;
    if (t < NTOK) {
      #pragma unroll
      for (int seg = 0; seg < 4; ++seg) {
        *(v8h*)&sQ[t * 32 + seg * 8] = *(const v8h*)(qb + (size_t)t * (3 * DIMC) + seg * 8);
        *(v8h*)&sK[t * 32 + seg * 8] = *(const v8h*)(qb + (size_t)t * (3 * DIMC) + DIMC + seg * 8);
        v8h vv = *(const v8h*)(qb + (size_t)t * (3 * DIMC) + 2 * DIMC + seg * 8);
        #pragma unroll
        for (int j = 0; j < 8; ++j) sVt[(seg * 8 + j) * 64 + t] = vv[j];
      }
    } else {
      v8h z = {};
      #pragma unroll
      for (int seg = 0; seg < 4; ++seg) {
        *(v8h*)&sQ[t * 32 + seg * 8] = z;
        *(v8h*)&sK[t * 32 + seg * 8] = z;
      }
      #pragma unroll
      for (int d = 0; d < 32; ++d) sVt[d * 64 + t] = (_Float16)0.f;
    }
  }
  __syncthreads();

  // ---- S = Q * K^T  (K-dim = hd = 32, one wmma per 16x16 tile) ----
  #pragma unroll
  for (int mi = 0; mi < 2; ++mi) {
    const int mtile = wave * 2 + mi;
    const int mrow  = mtile * 16 + (lane & 15);
    const int kh    = (lane >> 4) * 16;
    v8h lo = *(const v8h*)&sQ[mrow * 32 + kh];
    v8h hi = *(const v8h*)&sQ[mrow * 32 + 8 + kh];
    v16h a = __builtin_shufflevector(lo, hi, 0,1,2,3,4,5,6,7,8,9,10,11,12,13,14,15);
    #pragma unroll
    for (int ni = 0; ni < 4; ++ni) {
      const int ncol = ni * 16 + (lane & 15);
      const int kb   = (lane >> 4) * 16;
      v16h b = *(const v16h*)&sK[ncol * 32 + kb];   // B[k][n] = K[n][k] contiguous
      v8f c = {};
      c = __builtin_amdgcn_wmma_f32_16x16x32_f16(false, a, false, b, (short)0, c, false, false);
      #pragma unroll
      for (int v = 0; v < 8; ++v)
        sS[(mtile * 16 + (lane >> 4) * 8 + v) * 64 + ni * 16 + (lane & 15)] = c[v];
    }
  }
  __syncthreads();

  // ---- softmax over j<49 with rel-pos bias + shift mask (row per thread) ----
  if (t < NTOK) {
    const int ih = t / WSZ, iw = t - ih * WSZ;
    const int wr = wi >> 3, wc = wi & 7;
    const int ri = swin_region(wr * WSZ + ih) * 3 + swin_region(wc * WSZ + iw);
    float mx = -1e30f;
    for (int j = 0; j < NTOK; ++j) {
      const int jh = j / WSZ, jw = j - jh * WSZ;
      const int ridx = (ih - jh + 6) * 13 + (iw - jw + 6);
      const int rj = swin_region(wr * WSZ + jh) * 3 + swin_region(wc * WSZ + jw);
      float l = sS[t * 64 + j] * 0.17677669529663687f        // 1/sqrt(32)
              + rel_table[ridx * NHEAD + h]
              + ((ri != rj) ? -100.0f : 0.0f);
      sS[t * 64 + j] = l;
      mx = fmaxf(mx, l);
    }
    float sum = 0.f;
    for (int j = 0; j < NTOK; ++j) {
      float e = __expf(sS[t * 64 + j] - mx);
      sS[t * 64 + j] = e; sum += e;
    }
    const float inv = 1.0f / sum;
    for (int j = 0; j < NTOK; ++j) sP[t * 64 + j] = (_Float16)(sS[t * 64 + j] * inv);
    for (int j = NTOK; j < 64; ++j) sP[t * 64 + j] = (_Float16)0.f;
  } else {
    for (int j = 0; j < 64; ++j) sP[t * 64 + j] = (_Float16)0.f;
  }
  __syncthreads();

  // ---- O = P * V  (K-dim = 64 in 2 steps of 32) ----
  v8f o[2][2];
  #pragma unroll
  for (int mi = 0; mi < 2; ++mi)
    #pragma unroll
    for (int ni = 0; ni < 2; ++ni) { v8f z = {}; o[mi][ni] = z; }
  #pragma unroll
  for (int s = 0; s < 2; ++s) {
    #pragma unroll
    for (int mi = 0; mi < 2; ++mi) {
      const int mtile = wave * 2 + mi;
      const int mrow  = mtile * 16 + (lane & 15);
      const int kh    = (lane >> 4) * 16;
      v8h lo = *(const v8h*)&sP[mrow * 64 + s * 32 + kh];
      v8h hi = *(const v8h*)&sP[mrow * 64 + s * 32 + 8 + kh];
      v16h a = __builtin_shufflevector(lo, hi, 0,1,2,3,4,5,6,7,8,9,10,11,12,13,14,15);
      #pragma unroll
      for (int ni = 0; ni < 2; ++ni) {
        const int ncol = ni * 16 + (lane & 15);
        const int kb   = s * 32 + (lane >> 4) * 16;
        v16h b = *(const v16h*)&sVt[ncol * 64 + kb];  // B[k][n] = Vt[n][k] contiguous
        o[mi][ni] = __builtin_amdgcn_wmma_f32_16x16x32_f16(
            false, a, false, b, (short)0, o[mi][ni], false, false);
      }
    }
  }
  // ---- store (rows >= 49 dropped) ----
  #pragma unroll
  for (int mi = 0; mi < 2; ++mi)
    #pragma unroll
    for (int ni = 0; ni < 2; ++ni)
      #pragma unroll
      for (int v = 0; v < 8; ++v) {
        const int row = wave * 32 + mi * 16 + (lane >> 4) * 8 + v;
        if (row < NTOK) {
          const int col = ni * 16 + (lane & 15);
          attnout[(wbase + row) * DIMC + h * HD + col] = (_Float16)o[mi][ni][v];
        }
      }
}

// --------------------------------- launch -----------------------------------

extern "C" void kernel_launch(void* const* d_in, const int* in_sizes, int n_in,
                              void* d_out, int out_size, void* d_ws, size_t ws_size,
                              hipStream_t stream) {
  (void)in_sizes; (void)n_in; (void)out_size; (void)ws_size;

  const float* x       = (const float*)d_in[0];
  const float* g1      = (const float*)d_in[1];
  const float* be1     = (const float*)d_in[2];
  const float* w_qkv   = (const float*)d_in[3];
  const float* b_qkv   = (const float*)d_in[4];
  const float* rel_tab = (const float*)d_in[5];
  const float* w_proj  = (const float*)d_in[6];
  const float* b_proj  = (const float*)d_in[7];
  const float* g2      = (const float*)d_in[8];
  const float* be2     = (const float*)d_in[9];
  const float* w1      = (const float*)d_in[10];
  const float* b1      = (const float*)d_in[11];
  const float* w2      = (const float*)d_in[12];
  const float* b2      = (const float*)d_in[13];
  float* out = (float*)d_out;

  // ---- workspace carve-out (sizes all 256B-aligned already) ----
  char* ws = (char*)d_ws;
  size_t off = 0;
  auto take = [&](size_t bytes) { size_t p = off; off += (bytes + 255) & ~(size_t)255; return p; };
  _Float16* wqkvT  = (_Float16*)(ws + take((size_t)DIMC * 3 * DIMC * 2));   // (1152,384)
  _Float16* wprojT = (_Float16*)(ws + take((size_t)DIMC * DIMC * 2));       // (384,384)
  _Float16* w1T    = (_Float16*)(ws + take((size_t)DIMC * NMLP * 2));       // (1536,384)
  _Float16* w2T    = (_Float16*)(ws + take((size_t)NMLP * DIMC * 2));       // (384,1536)
  _Float16* xw     = (_Float16*)(ws + take((size_t)TOK * DIMC * 2));        // 77 MB (reused as ln2)
  size_t qkv_off   = take((size_t)TOK * 3 * DIMC * 2);                      // 231 MB
  _Float16* qkvbuf = (_Float16*)(ws + qkv_off);
  _Float16* attnout= (_Float16*)(ws + take((size_t)TOK * DIMC * 2));        // 77 MB (follows qkv)
  float*    ywin   = (float*)(ws + take((size_t)TOK * DIMC * 4));           // 154 MB
  float*    x1     = (float*)(ws + take((size_t)TOK * DIMC * 4));           // 154 MB
  _Float16* ln2    = xw;                          // xw dead after QKV GEMM
  _Float16* h1     = (_Float16*)(ws + qkv_off);   // qkv+attnout region (308 MB) reused

  // 1) weights -> f16, transposed to (N,K) for async-friendly B tiles
  k_cvt_t_f16<<<(DIMC * 3 * DIMC + 255) / 256, 256, 0, stream>>>(w_qkv, wqkvT, DIMC, 3 * DIMC);
  k_cvt_t_f16<<<(DIMC * DIMC + 255) / 256, 256, 0, stream>>>(w_proj, wprojT, DIMC, DIMC);
  k_cvt_t_f16<<<(DIMC * NMLP + 255) / 256, 256, 0, stream>>>(w1, w1T, DIMC, NMLP);
  k_cvt_t_f16<<<(NMLP * DIMC + 255) / 256, 256, 0, stream>>>(w2, w2T, NMLP, DIMC);

  // 2) LN1 + roll + window partition -> xw (f16, windowed order)
  k_ln1_window<<<TOK / 8, 256, 0, stream>>>(x, g1, be1, xw);

  // 3) QKV GEMM: (100352 x 384) @ (384 x 1152) -> qkv f16
  k_gemm<false, false, false><<<dim3(3 * DIMC / 128, TOK / 128), 256, 0, stream>>>(
      xw, wqkvT, b_qkv, nullptr, (void*)qkvbuf, TOK, 3 * DIMC, DIMC);

  // 4) windowed attention (2048 windows x 12 heads)
  k_attn<<<(TOK / NTOK) * NHEAD, 64, 0, stream>>>(qkvbuf, rel_tab, attnout);

  // 5) output projection: (100352 x 384) @ (384 x 384) -> ywin f32 (windowed)
  k_gemm<true, false, false><<<dim3(DIMC / 128, TOK / 128), 256, 0, stream>>>(
      attnout, wprojT, b_proj, nullptr, (void*)ywin, TOK, DIMC, DIMC);

  // 6) window reverse + roll + residual -> x1 (f32); LN2 -> ln2 (f16)
  k_unwin_ln2<<<TOK / 8, 256, 0, stream>>>(ywin, x, g2, be2, x1, ln2);

  // 7) MLP1 + GELU: (100352 x 384) @ (384 x 1536) -> h1 f16
  k_gemm<false, true, false><<<dim3(NMLP / 128, TOK / 128), 256, 0, stream>>>(
      ln2, w1T, b1, nullptr, (void*)h1, TOK, NMLP, DIMC);

  // 8) MLP2 + residual: (100352 x 1536) @ (1536 x 384) + x1 -> d_out f32
  k_gemm<true, false, true><<<dim3(DIMC / 128, TOK / 128), 256, 0, stream>>>(
      h1, w2T, b2, x1, (void*)out, TOK, DIMC, NMLP);
}
